// LinearAttentionLayer_43550968382293
// MI455X (gfx1250) — compile-verified, matrix operand order checked by reading, not run on hardware
//
#include <hip/hip_runtime.h>
#include <hip/hip_bf16.h>

// ---------------------------------------------------------------------------
// Performer / FAVOR+ causal linear attention for MI455X (gfx1250, wave32).
//
// All matrix math on v_wmma_f32_16x16x32_bf16 (f32 accumulate).
// Pipeline:
//   0) pre-convert activations f32->bf16; pre-transpose+convert weights
//   1) Q/K/V projections (bf16 GEMM, double-buffered LDS, pure-copy staging)
//   2) key max over (L,M) per (b,h)
//   3) FAVOR feature maps (GEMM + exp epilogue)
//   4) chunked causal linear attention (chunk=64):
//        S_c = Kl_c^T V_c, z_c = sum Kl_c          (GEMM)
//        exclusive prefix over chunks               (f32 adds)
//        out = (tril(Q Kl^T) V + Q P)/(rowsum + Qz) (3x GEMM)
//   5) output projection (bf16 GEMM)
// ---------------------------------------------------------------------------

constexpr int BB = 2;
constexpr int LL = 4096;
constexpr int DMODEL = 1024;
constexpr int HH = 16;
constexpr int MM = 64;
constexpr int ROWS  = BB * LL;          // 8192
constexpr int FROWS = BB * LL * HH;     // 131072
constexpr int NCHUNK = LL / 64;         // 64
constexpr float DN = 0.35355339059327379f;   // M^(-1/4)
constexpr float DIAG_SCALE = 0.0625f;        // 0.5 * DN^2
constexpr float INV_SQRT_M = 0.125f;         // 1/sqrt(64)

typedef __attribute__((ext_vector_type(16))) __bf16 bf16x16;
typedef __attribute__((ext_vector_type(8)))  float  f32x8;

union FragU { unsigned int u[8]; bf16x16 v; };
union AccU  { f32x8 v; float f[8]; };

__device__ __forceinline__ f32x8 zero8() {
  f32x8 z = {0.f, 0.f, 0.f, 0.f, 0.f, 0.f, 0.f, 0.f};
  return z;
}

// Native conversion; backend selects v_cvt_pk_bf16_f32-class ops where present.
__device__ __forceinline__ unsigned short f2bf(float x) {
  union { __bf16 h; unsigned short u; } c;
  c.h = (__bf16)x;
  return c.u;
}
__device__ __forceinline__ float bf2f(unsigned short h) {
  union { unsigned int u; float f; } a; a.u = ((unsigned int)h) << 16;
  return a.f;
}

// Fragment load from LDS tile stored K-contiguous per row (row = M for A,
// row = N for B^T). K packing per CDNA5 ISA 16-bit 16x32 A layout:
//   vgpr i, lane-half hh: k = (i>=4?16:0) + hh*8 + (i&3)*2  (pair k,k+1)
// -> two contiguous 16B runs per lane; compiler merges into ds_load_b128 x2.
template <int PITCH>
__device__ __forceinline__ bf16x16 ldsFrag(const unsigned short* base, int row0,
                                           int k0, int lane) {
  const int r  = row0 + (lane & 15);
  const int kh = (lane >> 4) << 3;
  FragU f;
#pragma unroll
  for (int i = 0; i < 8; ++i) {
    const int k = k0 + ((i & 4) << 2) + kh + ((i & 3) << 1);
    f.u[i] = *(const unsigned int*)(base + r * PITCH + k);
  }
  return f.v;
}

__device__ __forceinline__ f32x8 wmma_bf16(bf16x16 a, bf16x16 b, f32x8 c) {
  return __builtin_amdgcn_wmma_f32_16x16x32_bf16(false, a, false, b, (short)0, c,
                                                 false, false);
}

// ---------------------------------------------------------------------------
// Elementwise f32 -> bf16 (8 elements / thread)
// ---------------------------------------------------------------------------
__global__ __launch_bounds__(256)
void cvt_bf16_kernel(const float* __restrict__ X, unsigned short* __restrict__ Y) {
  const unsigned i = (blockIdx.x * 256u + threadIdx.x) * 8u;
  const float4 v0 = *(const float4*)(X + i);
  const float4 v1 = *(const float4*)(X + i + 4);
  union { unsigned short s[8]; uint4 v; } o;
  o.s[0] = f2bf(v0.x); o.s[1] = f2bf(v0.y); o.s[2] = f2bf(v0.z); o.s[3] = f2bf(v0.w);
  o.s[4] = f2bf(v1.x); o.s[5] = f2bf(v1.y); o.s[6] = f2bf(v1.z); o.s[7] = f2bf(v1.w);
  *(uint4*)(Y + i) = o.v;
}

// ---------------------------------------------------------------------------
// Weight transpose + convert: W[K,N] f32 -> Wt[N,K] bf16 (32x32 LDS tiles)
// ---------------------------------------------------------------------------
__global__ __launch_bounds__(256)
void wtrans_kernel(const float* __restrict__ W, unsigned short* __restrict__ Wt) {
  __shared__ unsigned short tile[32][33];
  const int k0 = blockIdx.y * 32, n0 = blockIdx.x * 32;
  const int tx = threadIdx.x & 31, ty = threadIdx.x >> 5;  // ty 0..7
#pragma unroll
  for (int r = ty; r < 32; r += 8)
    tile[r][tx] = f2bf(W[(unsigned)(k0 + r) * DMODEL + n0 + tx]);
  __syncthreads();
#pragma unroll
  for (int r = ty; r < 32; r += 8)
    Wt[(unsigned)(n0 + r) * DMODEL + k0 + tx] = tile[tx][r];
}

// ---------------------------------------------------------------------------
// GEMM: C[M,N] = A[M,K] @ Bt[N,K]^T + bias[N]
// A, Bt bf16 row-major; OUT_BF16 selects f32 or bf16 output at compile time.
// 256 thr = 8 waves (4x2), tile 128x128, BK=32, double-buffered LDS,
// global->VGPR->LDS staging is a pure copy (no conversion in the k-loop).
// ---------------------------------------------------------------------------
template <int OUT_BF16>
__global__ __launch_bounds__(256)
void gemm_bf16_128x128(const unsigned short* __restrict__ A,
                       const unsigned short* __restrict__ Bt,
                       const float* __restrict__ bias,
                       void* __restrict__ Cout,
                       int Mdim, int Ndim, int Kdim) {
  __shared__ __align__(16) unsigned short LA[2][128][40];  // [buf][m][k]
  __shared__ __align__(16) unsigned short LB[2][128][40];  // [buf][n][k]
  const int tid  = threadIdx.x;
  const int lane = tid & 31;
  const int wave = tid >> 5;
  const int wm = (wave >> 1) * 32;
  const int wn = (wave & 1) * 64;
  const int bm = blockIdx.y * 128;
  const int bn = blockIdx.x * 128;

  f32x8 acc[2][4];
#pragma unroll
  for (int i = 0; i < 2; ++i)
#pragma unroll
    for (int j = 0; j < 4; ++j) acc[i][j] = zero8();

  const int srow = tid >> 1;          // 0..127
  const int soff = (tid & 1) << 4;    // 0 or 16 (bf16 elements)
  const unsigned short* pa = A  + (unsigned)(bm + srow) * Kdim + soff;
  const unsigned short* pb = Bt + (unsigned)(bn + srow) * Kdim + soff;

  // prologue: tile 0 -> LDS buffer 0
  {
    uint4 a0 = ((const uint4*)pa)[0], a1 = ((const uint4*)pa)[1];
    uint4 b0 = ((const uint4*)pb)[0], b1 = ((const uint4*)pb)[1];
    uint4* da = (uint4*)&LA[0][srow][soff];
    uint4* db = (uint4*)&LB[0][srow][soff];
    da[0] = a0; da[1] = a1;
    db[0] = b0; db[1] = b1;
  }
  __syncthreads();

  const int T = Kdim >> 5;
  for (int t = 0; t < T; ++t) {
    const int cur = t & 1;
    uint4 a0, a1, b0, b1;
    if (t + 1 < T) {  // stage next tile into registers while computing
      const unsigned ko = (unsigned)(t + 1) << 5;
      a0 = ((const uint4*)(pa + ko))[0]; a1 = ((const uint4*)(pa + ko))[1];
      b0 = ((const uint4*)(pb + ko))[0]; b1 = ((const uint4*)(pb + ko))[1];
      if (t + 2 < T) {
        __builtin_prefetch(pa + ko + 32, 0, 1);   // global_prefetch_b8
        __builtin_prefetch(pb + ko + 32, 0, 1);
      }
    }

    bf16x16 af[2], bfr[4];
#pragma unroll
    for (int i = 0; i < 2; ++i)
      af[i] = ldsFrag<40>(&LA[cur][0][0], wm + i * 16, 0, lane);
#pragma unroll
    for (int j = 0; j < 4; ++j)
      bfr[j] = ldsFrag<40>(&LB[cur][0][0], wn + j * 16, 0, lane);
#pragma unroll
    for (int i = 0; i < 2; ++i)
#pragma unroll
      for (int j = 0; j < 4; ++j) acc[i][j] = wmma_bf16(af[i], bfr[j], acc[i][j]);

    if (t + 1 < T) {
      uint4* da = (uint4*)&LA[cur ^ 1][srow][soff];
      uint4* db = (uint4*)&LB[cur ^ 1][srow][soff];
      da[0] = a0; da[1] = a1;
      db[0] = b0; db[1] = b1;
      __syncthreads();
    }
  }

  const int nl = lane & 15;
  const int mh = (lane >> 4) << 3;
#pragma unroll
  for (int i = 0; i < 2; ++i)
#pragma unroll
    for (int j = 0; j < 4; ++j) {
      AccU a; a.v = acc[i][j];
      const int n = bn + wn + j * 16 + nl;
      const float bv = bias[n];
      const unsigned base = (unsigned)(bm + wm + i * 16 + mh) * (unsigned)Ndim + n;
      if constexpr (OUT_BF16) {
        unsigned short* cp = (unsigned short*)Cout + base;
#pragma unroll
        for (int q = 0; q < 8; ++q) { cp[0] = f2bf(a.f[q] + bv); cp += Ndim; }
      } else {
        float* cp = (float*)Cout + base;
#pragma unroll
        for (int q = 0; q < 8; ++q) { cp[0] = a.f[q] + bv; cp += Ndim; }
      }
    }
}

// ---------------------------------------------------------------------------
// Key max over (L, M) per (b,h): kmax[bh] = DN * max_{l,m} K[b,l,h,m]
// ---------------------------------------------------------------------------
__global__ __launch_bounds__(256)
void kmax_kernel(const float* __restrict__ K, float* __restrict__ kmax) {
  const int bh = blockIdx.x;
  const int b = bh >> 4, h = bh & 15;
  const int tid = threadIdx.x;
  float mx = -3.0e38f;
  for (int l = tid; l < LL; l += 256) {
    const float4* src = (const float4*)(K + ((unsigned)(b * LL + l) * HH + h) * 64u);
#pragma unroll
    for (int i = 0; i < 16; ++i) {
      float4 v = src[i];
      mx = fmaxf(mx, fmaxf(fmaxf(v.x, v.y), fmaxf(v.z, v.w)));
    }
  }
  __shared__ float red[256];
  red[tid] = mx;
  __syncthreads();
  for (int s = 128; s > 0; s >>= 1) {
    if (tid < s) red[tid] = fmaxf(red[tid], red[tid + s]);
    __syncthreads();
  }
  if (tid == 0) kmax[bh] = red[0] * DN;
}

// ---------------------------------------------------------------------------
// FAVOR feature map: Out = 1/sqrt(M) * exp(x_s @ Wf^T - diag - mx + 1e-8)
// x_s = DN*x; diag = 0.5*DN^2*||x_s||^2; mx = rowmax(x_s) (query) or kmax (key)
// ---------------------------------------------------------------------------
__global__ __launch_bounds__(128)
void favor_kernel(const float* __restrict__ X, const float* __restrict__ Wf,
                  const float* __restrict__ kmax, unsigned short* __restrict__ Out,
                  int is_query) {
  __shared__ __align__(16) unsigned short LX[128][72];  // [row][m] scaled bf16
  __shared__ __align__(16) unsigned short LW[64][72];   // [d][m] == Wf row-major
  __shared__ float diagv[128];
  __shared__ float mxv[128];
  const int tid = threadIdx.x, lane = tid & 31, wave = tid >> 5;
  const unsigned r0 = blockIdx.x * 128u;

  {
    const int d = tid >> 1, mh = (tid & 1) << 5;
    const float4* src = (const float4*)(Wf + d * 64 + mh);
#pragma unroll
    for (int i = 0; i < 8; ++i) {
      float4 v = src[i];
      LW[d][mh + 4 * i + 0] = f2bf(v.x);
      LW[d][mh + 4 * i + 1] = f2bf(v.y);
      LW[d][mh + 4 * i + 2] = f2bf(v.z);
      LW[d][mh + 4 * i + 3] = f2bf(v.w);
    }
  }
  {
    const float4* src = (const float4*)(X + (size_t)(r0 + tid) * 64);
    float ss = 0.f, mx = -3.0e38f;
#pragma unroll
    for (int i = 0; i < 16; ++i) {
      float4 v = src[i];
      float a = v.x * DN, b = v.y * DN, c = v.z * DN, d = v.w * DN;
      ss += a * a + b * b + c * c + d * d;
      mx = fmaxf(mx, fmaxf(fmaxf(a, b), fmaxf(c, d)));
      LX[tid][4 * i + 0] = f2bf(a);
      LX[tid][4 * i + 1] = f2bf(b);
      LX[tid][4 * i + 2] = f2bf(c);
      LX[tid][4 * i + 3] = f2bf(d);
    }
    diagv[tid] = ss * DIAG_SCALE;
    if (is_query) {
      mxv[tid] = mx;
    } else {
      const unsigned row = r0 + tid;           // row = (b*L + l)*H + h
      const int h = (int)(row & (HH - 1));
      const int b = (int)(row >> 16);          // / (L*H)
      mxv[tid] = kmax[b * HH + h];
    }
  }
  __syncthreads();

  f32x8 acc[2][4];
#pragma unroll
  for (int i = 0; i < 2; ++i)
#pragma unroll
    for (int j = 0; j < 4; ++j) acc[i][j] = zero8();

#pragma unroll
  for (int ksx = 0; ksx < 2; ++ksx) {
    bf16x16 af[2], bw[4];
#pragma unroll
    for (int i = 0; i < 2; ++i)
      af[i] = ldsFrag<72>(&LX[0][0], (wave * 2 + i) * 16, ksx * 32, lane);
#pragma unroll
    for (int j = 0; j < 4; ++j) bw[j] = ldsFrag<72>(&LW[0][0], j * 16, ksx * 32, lane);
#pragma unroll
    for (int i = 0; i < 2; ++i)
#pragma unroll
      for (int j = 0; j < 4; ++j) acc[i][j] = wmma_bf16(af[i], bw[j], acc[i][j]);
  }

  const int nl = lane & 15, mh2 = (lane >> 4) << 3;
#pragma unroll
  for (int i = 0; i < 2; ++i)
#pragma unroll
    for (int j = 0; j < 4; ++j) {
      AccU a; a.v = acc[i][j];
      const int d = j * 16 + nl;
#pragma unroll
      for (int q = 0; q < 8; ++q) {
        const int rl = (wave * 2 + i) * 16 + mh2 + q;
        const float e = INV_SQRT_M * __expf(a.f[q] - diagv[rl] - mxv[rl] + 1e-8f);
        Out[(unsigned)(r0 + rl) * 64u + d] = f2bf(e);
      }
    }
}

// ---------------------------------------------------------------------------
// Per-chunk stats: S_c[m][d] = sum_l Kl[l][m]*V[l][d]; z_c[m] = sum_l Kl[l][m]
// ---------------------------------------------------------------------------
__global__ __launch_bounds__(128)
void chunk_stats_kernel(const unsigned short* __restrict__ Kl,
                        const unsigned short* __restrict__ Vb,
                        float* __restrict__ S, float* __restrict__ Z) {
  const int blk = blockIdx.x;
  const int bh = blk >> 6, c = blk & 63;
  const int b = bh >> 4, h = bh & 15;
  const int tid = threadIdx.x, lane = tid & 31, wave = tid >> 5;
  __shared__ __align__(16) unsigned short LA[64][72];  // [m][l] = Kl[l][m]
  __shared__ __align__(16) unsigned short LB[64][72];  // [d][l] = V[l][d]
  {
    const int l = tid & 63, half = (tid >> 6) << 5;
    const unsigned row = (unsigned)(b * LL + c * 64 + l) * HH + h;
    const unsigned short* ks = Kl + row * 64u + half;
    const unsigned short* vs = Vb + row * 64u + half;
#pragma unroll
    for (int i = 0; i < 32; ++i) { LA[half + i][l] = ks[i]; LB[half + i][l] = vs[i]; }
  }
  __syncthreads();

  f32x8 acc[4];
#pragma unroll
  for (int j = 0; j < 4; ++j) acc[j] = zero8();
#pragma unroll
  for (int ksx = 0; ksx < 2; ++ksx) {
    bf16x16 af = ldsFrag<72>(&LA[0][0], wave * 16, ksx * 32, lane);
#pragma unroll
    for (int j = 0; j < 4; ++j) {
      bf16x16 bv = ldsFrag<72>(&LB[0][0], j * 16, ksx * 32, lane);
      acc[j] = wmma_bf16(af, bv, acc[j]);
    }
  }
  float* Sout = S + (unsigned)blk * 4096u;
  const int nl = lane & 15, mh = (lane >> 4) << 3;
#pragma unroll
  for (int j = 0; j < 4; ++j) {
    AccU a; a.v = acc[j];
    float* cp = Sout + (unsigned)(wave * 16 + mh) * 64u + j * 16 + nl;
#pragma unroll
    for (int q = 0; q < 8; ++q) { cp[0] = a.f[q]; cp += 64; }
  }
  if (tid < 64) {
    float s = 0.f;
#pragma unroll 8
    for (int l = 0; l < 64; ++l) s += bf2f(LA[tid][l]);
    Z[(unsigned)blk * 64u + tid] = s;
  }
}

// ---------------------------------------------------------------------------
// Exclusive prefix over chunks (in place). One block per (b,h).
// ---------------------------------------------------------------------------
__global__ __launch_bounds__(256)
void prefix_kernel(float* __restrict__ S, float* __restrict__ Z) {
  const int bh = blockIdx.x, tid = threadIdx.x;
  float run[16];
#pragma unroll
  for (int i = 0; i < 16; ++i) run[i] = 0.f;
  float zrun = 0.f;
  float* Sb = S + (unsigned)bh * NCHUNK * 4096u;
  float* Zb = Z + (unsigned)bh * NCHUNK * 64u;
  for (int c = 0; c < NCHUNK; ++c) {
    float* Sc = Sb + (unsigned)c * 4096u;
#pragma unroll
    for (int i = 0; i < 16; ++i) {
      const int e = i * 256 + tid;
      const float t = Sc[e];
      Sc[e] = run[i];
      run[i] += t;
    }
    if (tid < 64) {
      const float t = Zb[c * 64 + tid];
      Zb[c * 64 + tid] = zrun;
      zrun += t;
    }
  }
}

// ---------------------------------------------------------------------------
// Intra-chunk: Sm = tril(Q Kl^T); out = (Sm V + Q P) / (rowsum(Sm) + Q z)
// Output written bf16 (it is only the A operand of the final projection).
// ---------------------------------------------------------------------------
__global__ __launch_bounds__(128)
void intra_chunk_kernel(const unsigned short* __restrict__ Ql,
                        const unsigned short* __restrict__ Kl,
                        const unsigned short* __restrict__ Vb,
                        const float* __restrict__ P, const float* __restrict__ Zp,
                        unsigned short* __restrict__ Ob) {
  const int blk = blockIdx.x;
  const int bh = blk >> 6, c = blk & 63;
  const int b = bh >> 4, h = bh & 15;
  const int tid = threadIdx.x, lane = tid & 31, wave = tid >> 5;

  __shared__ __align__(16) unsigned short LQ[64][72];  // [l][m]
  __shared__ __align__(16) unsigned short LK[64][72];  // [j][m]
  __shared__ __align__(16) unsigned short LS[64][72];  // masked scores [l][j]
  __shared__ __align__(16) unsigned short LV[64][72];  // [d][l]
  __shared__ __align__(16) unsigned short LP[64][72];  // [d][m]
  __shared__ float zsh[64];
  __shared__ float den[64];

  {
    const int l = tid & 63, half = (tid >> 6) << 5;
    const unsigned row = (unsigned)(b * LL + c * 64 + l) * HH + h;
    const unsigned short* qs = Ql + row * 64u + half;
    const unsigned short* ks = Kl + row * 64u + half;
    const unsigned short* vs = Vb + row * 64u + half;
#pragma unroll
    for (int i = 0; i < 32; ++i) {
      LQ[l][half + i] = qs[i];
      LK[l][half + i] = ks[i];
      LV[half + i][l] = vs[i];
    }
    const float* ps = P + (unsigned)blk * 4096u + (unsigned)l * 64u + half;
#pragma unroll
    for (int i = 0; i < 32; ++i) LP[half + i][l] = f2bf(ps[i]);
    if (tid < 64) zsh[tid] = Zp[(unsigned)blk * 64u + tid];
  }
  __syncthreads();

  // GEMM1: scores = Q @ Kl^T  (wave w owns rows 16w..16w+15)
  f32x8 accS[4];
#pragma unroll
  for (int j = 0; j < 4; ++j) accS[j] = zero8();
#pragma unroll
  for (int ksx = 0; ksx < 2; ++ksx) {
    bf16x16 aq = ldsFrag<72>(&LQ[0][0], wave * 16, ksx * 32, lane);
#pragma unroll
    for (int j = 0; j < 4; ++j) {
      bf16x16 bk = ldsFrag<72>(&LK[0][0], j * 16, ksx * 32, lane);
      accS[j] = wmma_bf16(aq, bk, accS[j]);
    }
  }
  const int nl = lane & 15, mh = (lane >> 4) << 3;
#pragma unroll
  for (int j = 0; j < 4; ++j) {
    AccU a; a.v = accS[j];
    const int jc = j * 16 + nl;
#pragma unroll
    for (int q = 0; q < 8; ++q) {
      const int l = wave * 16 + mh + q;
      LS[l][jc] = f2bf(jc <= l ? a.f[q] : 0.f);   // causal (inclusive) mask
    }
  }
  __syncthreads();

  if (tid < 64) {  // denominator: rowsum(Sm) + Q.z
    float s = 0.f;
#pragma unroll 8
    for (int j = 0; j < 64; ++j) s += bf2f(LS[tid][j]);
#pragma unroll 8
    for (int m = 0; m < 64; ++m) s += bf2f(LQ[tid][m]) * zsh[m];
    den[tid] = s;
  }

  // GEMM2: Sm @ V  +  GEMM3: Q @ P  (accumulated into the same tiles)
  f32x8 acc[4];
#pragma unroll
  for (int j = 0; j < 4; ++j) acc[j] = zero8();
#pragma unroll
  for (int ksx = 0; ksx < 2; ++ksx) {
    bf16x16 as = ldsFrag<72>(&LS[0][0], wave * 16, ksx * 32, lane);
#pragma unroll
    for (int j = 0; j < 4; ++j) {
      bf16x16 bv = ldsFrag<72>(&LV[0][0], j * 16, ksx * 32, lane);
      acc[j] = wmma_bf16(as, bv, acc[j]);
    }
  }
#pragma unroll
  for (int ksx = 0; ksx < 2; ++ksx) {
    bf16x16 aq = ldsFrag<72>(&LQ[0][0], wave * 16, ksx * 32, lane);
#pragma unroll
    for (int j = 0; j < 4; ++j) {
      bf16x16 bp = ldsFrag<72>(&LP[0][0], j * 16, ksx * 32, lane);
      acc[j] = wmma_bf16(aq, bp, acc[j]);
    }
  }
  __syncthreads();

#pragma unroll
  for (int j = 0; j < 4; ++j) {
    AccU a; a.v = acc[j];
    const int d = j * 16 + nl;
#pragma unroll
    for (int q = 0; q < 8; ++q) {
      const int l = wave * 16 + mh + q;
      const unsigned row = (unsigned)(b * LL + c * 64 + l) * HH + h;
      Ob[row * 64u + d] = f2bf(a.f[q] / den[l]);
    }
  }
}

// ---------------------------------------------------------------------------
// Workspace layout (bytes)
// ---------------------------------------------------------------------------
constexpr size_t SZ_TOK_F = (size_t)ROWS * DMODEL * 4;   // 32 MB
constexpr size_t SZ_TOK_B = (size_t)ROWS * DMODEL * 2;   // 16 MB
constexpr size_t SZ_W_B   = (size_t)DMODEL * DMODEL * 2; // 2 MB
constexpr size_t SZ_F     = (size_t)FROWS * MM * 2;      // 16 MB
constexpr size_t SZ_S     = (size_t)BB * HH * NCHUNK * 64 * 64 * 4;
constexpr size_t SZ_Z     = (size_t)BB * HH * NCHUNK * 64 * 4;

constexpr size_t OFF_XQ  = 0;
constexpr size_t OFF_XK  = OFF_XQ + SZ_TOK_B;
constexpr size_t OFF_XV  = OFF_XK + SZ_TOK_B;
constexpr size_t OFF_WTQ = OFF_XV + SZ_TOK_B;
constexpr size_t OFF_WTK = OFF_WTQ + SZ_W_B;
constexpr size_t OFF_WTV = OFF_WTK + SZ_W_B;
constexpr size_t OFF_WTO = OFF_WTV + SZ_W_B;
constexpr size_t OFF_Q   = OFF_WTO + SZ_W_B;   // f32
constexpr size_t OFF_K   = OFF_Q + SZ_TOK_F;   // f32
constexpr size_t OFF_VB  = OFF_K + SZ_TOK_F;   // bf16
constexpr size_t OFF_QF  = OFF_VB + SZ_TOK_B;  // bf16 features
constexpr size_t OFF_KF  = OFF_QF + SZ_F;
constexpr size_t OFF_S   = OFF_KF + SZ_F;
constexpr size_t OFF_Z   = OFF_S + SZ_S;
constexpr size_t OFF_OM  = OFF_Z + SZ_Z;       // bf16 attention out
constexpr size_t OFF_KM  = OFF_OM + SZ_TOK_B;

extern "C" void kernel_launch(void* const* d_in, const int* in_sizes, int n_in,
                              void* d_out, int out_size, void* d_ws, size_t ws_size,
                              hipStream_t stream) {
  (void)in_sizes; (void)n_in; (void)out_size; (void)ws_size;
  const float* queries = (const float*)d_in[0];
  const float* keys    = (const float*)d_in[1];
  const float* values  = (const float*)d_in[2];
  const float* Wq = (const float*)d_in[3];
  const float* bq = (const float*)d_in[4];
  const float* Wk = (const float*)d_in[5];
  const float* bk = (const float*)d_in[6];
  const float* Wv = (const float*)d_in[7];
  const float* bv = (const float*)d_in[8];
  const float* Wo = (const float*)d_in[9];
  const float* bo = (const float*)d_in[10];
  const float* Wfq = (const float*)d_in[11];
  const float* Wfk = (const float*)d_in[12];

  char* ws = (char*)d_ws;
  unsigned short* Xq  = (unsigned short*)(ws + OFF_XQ);
  unsigned short* Xk  = (unsigned short*)(ws + OFF_XK);
  unsigned short* Xv  = (unsigned short*)(ws + OFF_XV);
  unsigned short* WtQ = (unsigned short*)(ws + OFF_WTQ);
  unsigned short* WtK = (unsigned short*)(ws + OFF_WTK);
  unsigned short* WtV = (unsigned short*)(ws + OFF_WTV);
  unsigned short* WtO = (unsigned short*)(ws + OFF_WTO);
  float* Q  = (float*)(ws + OFF_Q);
  float* Kp = (float*)(ws + OFF_K);
  unsigned short* Vb = (unsigned short*)(ws + OFF_VB);
  unsigned short* Qf = (unsigned short*)(ws + OFF_QF);
  unsigned short* Kf = (unsigned short*)(ws + OFF_KF);
  float* Sm = (float*)(ws + OFF_S);
  float* Zm = (float*)(ws + OFF_Z);
  unsigned short* Omb = (unsigned short*)(ws + OFF_OM);
  float* Km = (float*)(ws + OFF_KM);

  const int nTok = ROWS * DMODEL;               // 8388608
  const dim3 gcvt(nTok / (256 * 8));            // 4096
  const dim3 gwt(DMODEL / 32, DMODEL / 32);     // 32x32
  const dim3 ggemm(DMODEL / 128, ROWS / 128);   // 8x64

  // 0) precision prep: activations -> bf16, weights -> bf16 transposed
  cvt_bf16_kernel<<<gcvt, 256, 0, stream>>>(queries, Xq);
  cvt_bf16_kernel<<<gcvt, 256, 0, stream>>>(keys,    Xk);
  cvt_bf16_kernel<<<gcvt, 256, 0, stream>>>(values,  Xv);
  wtrans_kernel<<<gwt, 256, 0, stream>>>(Wq, WtQ);
  wtrans_kernel<<<gwt, 256, 0, stream>>>(Wk, WtK);
  wtrans_kernel<<<gwt, 256, 0, stream>>>(Wv, WtV);
  wtrans_kernel<<<gwt, 256, 0, stream>>>(Wo, WtO);

  // 1) input projections (Q,K -> f32 for feature-map stats; V -> bf16 direct)
  gemm_bf16_128x128<0><<<ggemm, 256, 0, stream>>>(Xq, WtQ, bq, Q,  ROWS, DMODEL, DMODEL);
  gemm_bf16_128x128<0><<<ggemm, 256, 0, stream>>>(Xk, WtK, bk, Kp, ROWS, DMODEL, DMODEL);
  gemm_bf16_128x128<1><<<ggemm, 256, 0, stream>>>(Xv, WtV, bv, Vb, ROWS, DMODEL, DMODEL);

  // 2) global key max per (b,h)
  kmax_kernel<<<BB * HH, 256, 0, stream>>>(Kp, Km);

  // 3) FAVOR feature maps
  favor_kernel<<<FROWS / 128, 128, 0, stream>>>(Q,  Wfq, nullptr, Qf, 1);
  favor_kernel<<<FROWS / 128, 128, 0, stream>>>(Kp, Wfk, Km,      Kf, 0);

  // 4) chunked causal linear attention
  chunk_stats_kernel<<<BB * HH * NCHUNK, 128, 0, stream>>>(Kf, Vb, Sm, Zm);
  prefix_kernel<<<BB * HH, 256, 0, stream>>>(Sm, Zm);
  intra_chunk_kernel<<<BB * HH * NCHUNK, 128, 0, stream>>>(Qf, Kf, Vb, Sm, Zm, Omb);

  // 5) output projection
  gemm_bf16_128x128<0><<<ggemm, 256, 0, stream>>>(Omb, WtO, bo, (float*)d_out,
                                                  ROWS, DMODEL, DMODEL);
}
